// SpectralAnalyzer_55877524521379
// MI455X (gfx1250) — matrix operand order repaired and checked by reading, not compiled, not run on hardware
//
#include <hip/hip_runtime.h>

#define PI_F 3.14159265358979323846f

typedef __attribute__((ext_vector_type(16))) _Float16 v16h;
typedef __attribute__((ext_vector_type(8)))  _Float16 v8h;
typedef __attribute__((ext_vector_type(8)))  float    v8f;

union V16 { v16h v; v8h h[2]; };

static constexpr int NB = 16;    // batch
static constexpr int NC = 3;     // input channels
static constexpr int HH = 256;
static constexpr int WW = 256;
static constexpr int WP = WW + 2;  // halo-padded row width for conv2 input
static constexpr int WR = 129;   // rfft width
static constexpr int NFREQ = HH * WR;   // 33024
static constexpr int NRAD = 32;
static constexpr int NAZ  = 8;

// ---------------------------------------------------------------------------
// helpers
// ---------------------------------------------------------------------------
__device__ __forceinline__ float n2n4(float v) {          // nan->0, inf->±1e4
  if (isnan(v)) return 0.f;
  if (isinf(v)) return v > 0.f ? 1e4f : -1e4f;
  return v;
}

// ---------------------------------------------------------------------------
// 256-point radix-2 DIT FFT in LDS, 128 threads, input already bit-reversed
// ---------------------------------------------------------------------------
__device__ __forceinline__ void fft256(float2* buf, int tid) {
  #pragma unroll
  for (int s = 1; s <= 8; ++s) {
    int half = 1 << (s - 1);
    int grp  = tid >> (s - 1);
    int pos  = tid & (half - 1);
    int i0   = (grp << s) + pos;
    int i1   = i0 + half;
    float ang = -2.0f * PI_F * (float)pos / (float)(half << 1);
    float sw, cw;
    __sincosf(ang, &sw, &cw);
    float2 a = buf[i0], b = buf[i1];
    float tr = cw * b.x - sw * b.y;
    float ti = cw * b.y + sw * b.x;
    buf[i0] = make_float2(a.x + tr, a.y + ti);
    buf[i1] = make_float2(a.x - tr, a.y - ti);
    __syncthreads();
  }
}

// one block per row (b,c,h): clip input, rFFT over W, keep bins 0..128
__global__ __launch_bounds__(128) void fft_row_kernel(const float* __restrict__ x,
                                                      float2* __restrict__ freq) {
  __shared__ float2 buf[256];
  int row = blockIdx.x;                 // [0, NB*NC*HH)
  int tid = threadIdx.x;
  const float* src = x + (size_t)row * WW;
  for (int i = tid; i < 256; i += 128) {
    float v = fminf(fmaxf(src[i], -10.f), 10.f);
    buf[__brev((unsigned)i) >> 24] = make_float2(v, 0.f);
  }
  __syncthreads();
  fft256(buf, tid);
  float2* dst = freq + (size_t)row * WR;
  for (int i = tid; i < WR; i += 128) dst[i] = buf[i];
}

// one block per column (b,c,x): complex FFT over H, in place
__global__ __launch_bounds__(128) void fft_col_kernel(float2* __restrict__ freq) {
  __shared__ float2 buf[256];
  int col = blockIdx.x;                 // [0, NB*NC*WR)
  int bc = col / WR, xc = col % WR;
  int tid = threadIdx.x;
  float2* base = freq + (size_t)bc * HH * WR + xc;
  for (int i = tid; i < 256; i += 128)
    buf[__brev((unsigned)i) >> 24] = base[(size_t)i * WR];
  __syncthreads();
  fft256(buf, tid);
  for (int i = tid; i < 256; i += 128) base[(size_t)i * WR] = buf[i];
}

// ---------------------------------------------------------------------------
// radial / azimuthal statistics : one block per (b,c)
// ---------------------------------------------------------------------------
__global__ __launch_bounds__(256) void stats_kernel(const float2* __restrict__ freq,
                                                    float* __restrict__ radial,
                                                    float* __restrict__ az) {
  __shared__ float rsum[NRAD], rcnt[NRAD], s1[NAZ], s2[NAZ], acnt[NAZ];
  int bc = blockIdx.x, tid = threadIdx.x;
  if (tid < NRAD) { rsum[tid] = 0.f; rcnt[tid] = 0.f; }
  if (tid < NAZ)  { s1[tid] = 0.f; s2[tid] = 0.f; acnt[tid] = 0.f; }
  __syncthreads();
  const float2* base = freq + (size_t)bc * NFREQ;
  const float rmax = sqrtf(0.5f);
  for (int n = tid; n < NFREQ; n += 256) {
    int y = n / WR, x = n % WR;
    float2 f = base[n];
    float mag = sqrtf(f.x * f.x + f.y * f.y);
    mag = fminf(fmaxf(mag, 1e-8f), 1e6f);
    float fy = (float)(y < 128 ? y : y - 256) * (1.0f / 256.0f);
    float fx = (float)x * (1.0f / 256.0f);
    float radius = sqrtf(fx * fx + fy * fy);
    int rb = (int)(radius / (rmax + 1e-8f) * (float)(NRAD - 1));
    rb = min(max(rb, 0), NRAD - 1);
    float ang = atan2f(fy, fx + 1e-8f);
    int ab = (int)((ang + PI_F) * ((float)NAZ / (2.0f * PI_F)));
    ab = min(max(ab, 0), NAZ - 1);
    atomicAdd(&rsum[rb], mag);
    atomicAdd(&rcnt[rb], 1.0f);
    atomicAdd(&s1[ab], mag);
    atomicAdd(&s2[ab], mag * mag);
    atomicAdd(&acnt[ab], 1.0f);
  }
  __syncthreads();
  int b = bc / NC, c = bc % NC;
  if (tid < NRAD) {
    float v = rsum[tid] / fmaxf(rcnt[tid], 1.0f);
    radial[bc * NRAD + tid] = fminf(fmaxf(v, 0.f), 1e6f);
  }
  if (tid < NAZ) {
    float cnt  = acnt[tid];
    float mean = s1[tid] / cnt;
    float var  = (s2[tid] - cnt * mean * mean) / fmaxf(cnt - 1.0f, 1.0f);
    float sd   = sqrtf(fmaxf(var, 0.f));
    az[b * (NAZ * 2 * NC) + tid * (2 * NC) + c]      = n2n4(mean);
    az[b * (NAZ * 2 * NC) + tid * (2 * NC) + NC + c] = n2n4(sd);
  }
}

// ---------------------------------------------------------------------------
// bilinear upsample [*,H,129] -> [*,H,W] of mag_log and phase (fp16 out)
// H dim: identity under half-pixel scale==1
// ---------------------------------------------------------------------------
__global__ __launch_bounds__(256) void upsample_kernel(const float2* __restrict__ freq,
                                                       _Float16* __restrict__ mag_sp,
                                                       _Float16* __restrict__ ph_sp) {
  int gid = blockIdx.x * 256 + threadIdx.x;    // NB*NC*HH*WW
  int x  = gid & 255;
  int y  = (gid >> 8) & 255;
  int bc = gid >> 16;
  float sx = ((float)x + 0.5f) * (129.0f / 256.0f) - 0.5f;
  float fx0 = floorf(sx);
  float w = sx - fx0;
  int x0 = (int)fx0;
  int x0c = max(x0, 0);
  int x1c = min(x0 + 1, WR - 1);
  const float2* rowp = freq + ((size_t)bc * HH + y) * WR;
  float2 f0 = rowp[x0c], f1 = rowp[x1c];
  auto maglog = [](float2 f) {
    float m = sqrtf(f.x * f.x + f.y * f.y);
    m = fminf(fmaxf(m, 1e-8f), 1e6f);
    return fminf(fmaxf(log1pf(m), -20.f), 20.f);
  };
  auto phasef = [](float2 f) {
    float p = atan2f(f.y, f.x) * (1.0f / PI_F);
    return fminf(fmaxf(p, -1.f), 1.f);
  };
  mag_sp[gid] = (_Float16)(maglog(f0) * (1.f - w) + maglog(f1) * w);
  ph_sp[gid]  = (_Float16)(phasef(f0) * (1.f - w) + phasef(f1) * w);
}

// ---------------------------------------------------------------------------
// conv1 (Cin=3, 3x3 SAME) + bias + BN + ReLU -> halo-padded NHWC fp16
// Direct VALU: K=27 is too small/ragged for WMMA; only ~5 GF total.
// Output layout: [B, H, W+2, COUT]; real pixels at x+1, halo cols zeroed
// separately so conv2 needs no per-lane x bounds checks.
// ---------------------------------------------------------------------------
template <int COUT>
__global__ __launch_bounds__(256) void conv1_kernel(const _Float16* __restrict__ xin,   // [B,3,H,W] fp16
                                                    const float* __restrict__ wgt,      // [COUT,3,3,3]
                                                    const float* __restrict__ bias,
                                                    const float* __restrict__ bn,       // [4,COUT]
                                                    _Float16* __restrict__ xact) {      // [B,H,W+2,COUT]
  int gid = blockIdx.x * 256 + threadIdx.x;
  int cout = gid % COUT;
  int p = gid / COUT;
  int x = p % WW;
  int y = (p / WW) % HH;
  int b = p / (WW * HH);
  float acc = bias[cout];
  #pragma unroll
  for (int cin = 0; cin < 3; ++cin) {
    #pragma unroll
    for (int ky = 0; ky < 3; ++ky) {
      int ys = y + ky - 1;
      if (ys < 0 || ys >= HH) continue;
      #pragma unroll
      for (int kx = 0; kx < 3; ++kx) {
        int xs = x + kx - 1;
        if (xs < 0 || xs >= WW) continue;
        float v = (float)xin[(((size_t)(b * 3 + cin)) * HH + ys) * WW + xs];
        acc += v * wgt[((cout * 3 + cin) * 3 + ky) * 3 + kx];
      }
    }
  }
  float g = bn[cout], be = bn[COUT + cout], m = bn[2 * COUT + cout], vv = bn[3 * COUT + cout];
  acc = (acc - m) * (g * rsqrtf(vv + 1e-5f)) + be;
  acc = fmaxf(acc, 0.f);
  xact[(((size_t)b * HH + y) * WP + (x + 1)) * COUT + cout] = (_Float16)acc;
}

// zero the two halo columns (x=0 and x=W+1) of a halo-padded NHWC buffer
template <int CIN>
__global__ void halo_zero_kernel(_Float16* __restrict__ xact) {
  int gid = blockIdx.x * 256 + threadIdx.x;
  int c = gid % CIN;
  int side = (gid / CIN) & 1;
  int row = gid / (2 * CIN);            // b*HH + y
  if (row >= NB * HH) return;
  int x = side ? (WW + 1) : 0;
  xact[((size_t)row * WP + x) * CIN + c] = (_Float16)0.f;
}

// pack conv2 weights [COUT,CIN,3,3] f32 -> [COUT, tap(9), CIN] fp16
template <int CIN, int COUT>
__global__ void pack_w_kernel(const float* __restrict__ w, _Float16* __restrict__ wpack) {
  int gid = blockIdx.x * 256 + threadIdx.x;
  if (gid >= COUT * 9 * CIN) return;
  int cin = gid % CIN;
  int tap = (gid / CIN) % 9;
  int cout = gid / (9 * CIN);
  wpack[gid] = (_Float16)w[(cout * CIN + cin) * 9 + tap];
}

// ---------------------------------------------------------------------------
// conv2 implicit GEMM with V_WMMA_F32_16X16X32_F16, fused bias+BN+ReLU+pool
//   A[M=cout][K] = packed weights   B[K][N=pixel] = halo-padded NHWC acts
//   K = tap*CIN + cin, stepped by 32.  Per-lane operand layouts per ISA 7.12.2:
//   A: lane<16 -> M=lane, halves = K{kb..kb+7, kb+16..23}; lane>=16 shifted 8/24
//   B: lane<16 -> N=lane, halves = K{kb..kb+7, kb+16..23}; lane>=16 shifted 8/24
//   C: vgpr r  -> M = r + (lane>=16 ? 8 : 0), N = lane%16
//   x-padding comes from the zero halo; y-padding is a *uniform* tap-row skip
//   (adding a zero row == skipping its wmma), so no EXEC masking in the loop.
// ---------------------------------------------------------------------------
template <int CIN, int COUT, int WM, int WN>
__global__ __launch_bounds__(256) void conv2_wmma_kernel(const _Float16* __restrict__ xact,  // [B,H,W+2,CIN]
                                                         const _Float16* __restrict__ wpack, // [COUT,9,CIN]
                                                         const float* __restrict__ bias,
                                                         const float* __restrict__ bn,       // [4,COUT]
                                                         float* __restrict__ pool) {         // [B,COUT] sums
  __shared__ float spool[COUT];
  int tid = threadIdx.x;
  int wid = tid >> 5, lane = tid & 31;
  int mwave = wid % WM, nwave = wid / WM;
  int b = blockIdx.z, y = blockIdx.y;
  int x_base = blockIdx.x * (WN * 16) + nwave * 16;
  int cout_base = mwave * 16;
  int lh = lane & 15;
  bool hi = lane >= 16;

  if (tid < COUT) spool[tid] = 0.f;
  __syncthreads();

  v8f acc = {};
  const _Float16* wrow = wpack + (size_t)(cout_base + lh) * 9 * CIN + (hi ? 8 : 0);
  int px = x_base + lh;                 // pixel x; halo index = px + kx
  int c_off = hi ? 8 : 0;
  const _Float16* xb = xact + (size_t)b * HH * WP * CIN + c_off;

  #pragma unroll
  for (int ky = 0; ky < 3; ++ky) {
    int ys = y + ky - 1;
    if (ys < 0 || ys >= HH) continue;   // uniform scalar branch == zero row
    const _Float16* rowp = xb + (size_t)ys * WP * CIN;
    // pull the tile a few pixels ahead toward L0 (feature map is L2-resident)
    __builtin_prefetch(rowp + (size_t)(px + 16) * CIN, 0, 0);
    #pragma unroll
    for (int kx = 0; kx < 3; ++kx) {
      int tap = ky * 3 + kx;
      const _Float16* bp = rowp + (size_t)(px + kx) * CIN;
      #pragma unroll
      for (int ks = 0; ks < CIN / 32; ++ks) {
        int kb = tap * CIN + ks * 32;
        V16 ua, ub;
        ua.h[0] = *(const v8h*)(wrow + kb);        // K kb+off .. +7
        ua.h[1] = *(const v8h*)(wrow + kb + 16);   // K kb+off+16 .. +23
        ub.h[0] = *(const v8h*)(bp + ks * 32);
        ub.h[1] = *(const v8h*)(bp + ks * 32 + 16);
        acc = __builtin_amdgcn_wmma_f32_16x16x32_f16(false, ua.v, false, ub.v,
                                                     (short)0, acc, false, false);
      }
    }
  }

  // fused bias + BN + ReLU, reduce over the 16 pixels (N) per half-wave
  #pragma unroll
  for (int r = 0; r < 8; ++r) {
    int cout = cout_base + r + (hi ? 8 : 0);
    float g = bn[cout], be = bn[COUT + cout], m = bn[2 * COUT + cout], vv = bn[3 * COUT + cout];
    float val = acc[r] + bias[cout];
    val = (val - m) * (g * rsqrtf(vv + 1e-5f)) + be;
    val = fmaxf(val, 0.f);
    #pragma unroll
    for (int off = 1; off < 16; off <<= 1) val += __shfl_xor(val, off, 32);
    if (lh == 0) atomicAdd(&spool[cout], val);
  }
  __syncthreads();
  if (tid < COUT) atomicAdd(&pool[b * COUT + tid], spool[tid]);
}

// ---------------------------------------------------------------------------
// radial branch: conv1d(3->32,k3) BN ReLU, conv1d(32->32,k3) BN ReLU, mean
// ---------------------------------------------------------------------------
__global__ __launch_bounds__(128) void radial_kernel(const float* __restrict__ radial, // [B,3,32]
                                                     const float* __restrict__ rw1, const float* __restrict__ rb1,
                                                     const float* __restrict__ rbn1,
                                                     const float* __restrict__ rw2, const float* __restrict__ rb2,
                                                     const float* __restrict__ rbn2,
                                                     float* __restrict__ rf) {         // [B,32]
  __shared__ float rin[3][32];
  __shared__ float f1[32][32];
  __shared__ float f2sum[32];
  int b = blockIdx.x, tid = threadIdx.x;
  if (tid < 96) rin[tid / 32][tid % 32] = radial[b * 96 + tid];
  if (tid < 32) f2sum[tid] = 0.f;
  __syncthreads();
  for (int i = tid; i < 32 * 32; i += 128) {
    int o = i / 32, l = i % 32;
    float acc = rb1[o];
    #pragma unroll
    for (int c = 0; c < 3; ++c)
      #pragma unroll
      for (int t = 0; t < 3; ++t) {
        int ls = l + t - 1;
        if (ls >= 0 && ls < 32) acc += rw1[(o * 3 + c) * 3 + t] * rin[c][ls];
      }
    float g = rbn1[o], be = rbn1[32 + o], m = rbn1[64 + o], v = rbn1[96 + o];
    acc = (acc - m) * (g * rsqrtf(v + 1e-5f)) + be;
    f1[o][l] = fmaxf(acc, 0.f);
  }
  __syncthreads();
  for (int i = tid; i < 32 * 32; i += 128) {
    int o = i / 32, l = i % 32;
    float acc = rb2[o];
    for (int ci = 0; ci < 32; ++ci)
      #pragma unroll
      for (int t = 0; t < 3; ++t) {
        int ls = l + t - 1;
        if (ls >= 0 && ls < 32) acc += rw2[(o * 32 + ci) * 3 + t] * f1[ci][ls];
      }
    float g = rbn2[o], be = rbn2[32 + o], m = rbn2[64 + o], v = rbn2[96 + o];
    acc = (acc - m) * (g * rsqrtf(v + 1e-5f)) + be;
    atomicAdd(&f2sum[o], fmaxf(acc, 0.f));
  }
  __syncthreads();
  if (tid < 32) rf[b * 32 + tid] = f2sum[tid] * (1.0f / 32.0f);
}

// ---------------------------------------------------------------------------
// head: concat -> linear(176->256) -> LayerNorm -> ReLU -> linear(256->128)
// one block per batch, 256 threads
// ---------------------------------------------------------------------------
__global__ __launch_bounds__(256) void head_kernel(const float* __restrict__ pool_mag, // [B,64] sums
                                                   const float* __restrict__ pool_ph,  // [B,32] sums
                                                   const float* __restrict__ rf,       // [B,32]
                                                   const float* __restrict__ az,       // [B,48]
                                                   const float* __restrict__ lw1, const float* __restrict__ lb1,
                                                   const float* __restrict__ lng, const float* __restrict__ lnb,
                                                   const float* __restrict__ lw2, const float* __restrict__ lb2,
                                                   float* __restrict__ out) {          // [B,128]
  __shared__ float comb[176];
  __shared__ float hbuf[256];
  __shared__ float red[256];
  int b = blockIdx.x, tid = threadIdx.x;
  const float inv_hw = 1.0f / (float)(HH * WW);
  if (tid < 64)       comb[tid] = n2n4(pool_mag[b * 64 + tid] * inv_hw);
  else if (tid < 96)  comb[tid] = n2n4(pool_ph[b * 32 + tid - 64] * inv_hw);
  else if (tid < 128) comb[tid] = n2n4(rf[b * 32 + tid - 96]);
  else if (tid < 176) comb[tid] = n2n4(az[b * 48 + tid - 128]);
  __syncthreads();

  float h = lb1[tid];
  for (int k = 0; k < 176; ++k) h += lw1[tid * 176 + k] * comb[k];

  red[tid] = h;
  __syncthreads();
  for (int off = 128; off > 0; off >>= 1) {
    if (tid < off) red[tid] += red[tid + off];
    __syncthreads();
  }
  float mu = red[0] * (1.0f / 256.0f);
  __syncthreads();
  red[tid] = (h - mu) * (h - mu);
  __syncthreads();
  for (int off = 128; off > 0; off >>= 1) {
    if (tid < off) red[tid] += red[tid + off];
    __syncthreads();
  }
  float var = red[0] * (1.0f / 256.0f);
  float hn = (h - mu) * rsqrtf(var + 1e-5f) * lng[tid] + lnb[tid];
  hbuf[tid] = fmaxf(hn, 0.f);
  __syncthreads();

  if (tid < 128) {
    float o = lb2[tid];
    for (int k = 0; k < 256; ++k) o += lw2[tid * 256 + k] * hbuf[k];
    if (isnan(o)) o = 0.f;
    o = fminf(fmaxf(o, -100.f), 100.f);
    out[b * 128 + tid] = o;
  }
}

__global__ void zero_kernel(float* __restrict__ p, int n) {
  int i = blockIdx.x * 256 + threadIdx.x;
  if (i < n) p[i] = 0.f;
}

// ---------------------------------------------------------------------------
extern "C" void kernel_launch(void* const* d_in, const int* in_sizes, int n_in,
                              void* d_out, int out_size, void* d_ws, size_t ws_size,
                              hipStream_t stream) {
  (void)in_sizes; (void)n_in; (void)out_size; (void)ws_size;
  const float* x    = (const float*)d_in[0];
  const float* mw1  = (const float*)d_in[1];
  const float* mb1  = (const float*)d_in[2];
  const float* mbn1 = (const float*)d_in[3];
  const float* mw2  = (const float*)d_in[4];
  const float* mb2  = (const float*)d_in[5];
  const float* mbn2 = (const float*)d_in[6];
  const float* pw1  = (const float*)d_in[7];
  const float* pb1  = (const float*)d_in[8];
  const float* pbn1 = (const float*)d_in[9];
  const float* pw2  = (const float*)d_in[10];
  const float* pb2  = (const float*)d_in[11];
  const float* pbn2 = (const float*)d_in[12];
  const float* rw1  = (const float*)d_in[13];
  const float* rb1  = (const float*)d_in[14];
  const float* rbn1 = (const float*)d_in[15];
  const float* rw2  = (const float*)d_in[16];
  const float* rb2  = (const float*)d_in[17];
  const float* rbn2 = (const float*)d_in[18];
  const float* lw1  = (const float*)d_in[19];
  const float* lb1  = (const float*)d_in[20];
  const float* lng  = (const float*)d_in[21];
  const float* lnb  = (const float*)d_in[22];
  const float* lw2  = (const float*)d_in[23];
  const float* lb2  = (const float*)d_in[24];

  char* ws = (char*)d_ws;
  size_t off = 0;
  auto carve = [&](size_t bytes) -> void* {
    void* p = (void*)(ws + off);
    off += (bytes + 255) & ~(size_t)255;
    return p;
  };
  float2*   freq     = (float2*)  carve((size_t)NB * NC * HH * WR * sizeof(float2));   // 12.7 MB
  _Float16* mag_sp   = (_Float16*)carve((size_t)NB * NC * HH * WW * 2);                // 6.3 MB
  _Float16* ph_sp    = (_Float16*)carve((size_t)NB * NC * HH * WW * 2);                // 6.3 MB
  _Float16* xact_m   = (_Float16*)carve((size_t)NB * HH * WP * 64 * 2);                // 135 MB
  _Float16* xact_p   = (_Float16*)carve((size_t)NB * HH * WP * 32 * 2);                // 68 MB
  _Float16* wpack_m  = (_Float16*)carve((size_t)64 * 9 * 64 * 2);
  _Float16* wpack_p  = (_Float16*)carve((size_t)32 * 9 * 32 * 2);
  float*    radial   = (float*)   carve((size_t)NB * NC * NRAD * 4);
  float*    az       = (float*)   carve((size_t)NB * NAZ * 2 * NC * 4);
  float*    pool_mag = (float*)   carve((size_t)NB * 64 * 4);   // contiguous with pool_ph
  float*    pool_ph  = (float*)   carve((size_t)NB * 32 * 4);
  float*    rf       = (float*)   carve((size_t)NB * 32 * 4);

  // 1) zero pooling accumulators (pool_mag || pool_ph contiguous = 1536 floats)
  zero_kernel<<<(1536 + 255) / 256, 256, 0, stream>>>(pool_mag, 1536);

  // 2) zero conv2-input halo columns; pack conv2 weights fp16 [cout][tap][cin]
  halo_zero_kernel<64><<<(NB * HH * 2 * 64 + 255) / 256, 256, 0, stream>>>(xact_m);
  halo_zero_kernel<32><<<(NB * HH * 2 * 32 + 255) / 256, 256, 0, stream>>>(xact_p);
  pack_w_kernel<64, 64><<<(64 * 9 * 64 + 255) / 256, 256, 0, stream>>>(mw2, wpack_m);
  pack_w_kernel<32, 32><<<(32 * 9 * 32 + 255) / 256, 256, 0, stream>>>(pw2, wpack_p);

  // 3) rfft2 = row rFFT then column complex FFT (LDS radix-2)
  fft_row_kernel<<<NB * NC * HH, 128, 0, stream>>>(x, freq);
  fft_col_kernel<<<NB * NC * WR, 128, 0, stream>>>(freq);

  // 4) radial / azimuthal statistics
  stats_kernel<<<NB * NC, 256, 0, stream>>>(freq, radial, az);

  // 5) bilinear upsample mag_log & phase to fp16 spatial maps
  upsample_kernel<<<(NB * NC * HH * WW) / 256, 256, 0, stream>>>(freq, mag_sp, ph_sp);

  // 6) conv1 (+BN+ReLU) -> halo-padded NHWC fp16
  conv1_kernel<64><<<(NB * HH * WW * 64) / 256, 256, 0, stream>>>(mag_sp, mw1, mb1, mbn1, xact_m);
  conv1_kernel<32><<<(NB * HH * WW * 32) / 256, 256, 0, stream>>>(ph_sp, pw1, pb1, pbn1, xact_p);

  // 7) conv2 via WMMA implicit GEMM, fused BN+ReLU+global-mean-pool
  //    mag: M=64 (4 waves), N=32 px (2 waves), K=576 (18 wmma/tile)
  conv2_wmma_kernel<64, 64, 4, 2><<<dim3(WW / 32, HH, NB), 256, 0, stream>>>(
      xact_m, wpack_m, mb2, mbn2, pool_mag);
  //    phase: M=32 (2 waves), N=64 px (4 waves), K=288 (9 wmma/tile)
  conv2_wmma_kernel<32, 32, 2, 4><<<dim3(WW / 64, HH, NB), 256, 0, stream>>>(
      xact_p, wpack_p, pb2, pbn2, pool_ph);

  // 8) radial 1-D conv branch
  radial_kernel<<<NB, 128, 0, stream>>>(radial, rw1, rb1, rbn1, rw2, rb2, rbn2, rf);

  // 9) head MLP + LayerNorm
  head_kernel<<<NB, 256, 0, stream>>>(pool_mag, pool_ph, rf, az,
                                      lw1, lb1, lng, lnb, lw2, lb2, (float*)d_out);
}
